// GConvSE3_48902497632467
// MI455X (gfx1250) — matrix operand order repaired and checked by reading, not compiled, not run on hardware
//
#include <hip/hip_runtime.h>
#include <math.h>

typedef __attribute__((ext_vector_type(16))) _Float16 v16h;
typedef __attribute__((ext_vector_type(8)))  float    v8f;
typedef _Float16 h16;

#define EPS 1e-5f

struct Args {
  const float* h0; const float* h1;
  const float* r;  const float* edge_feat;
  const float* basis[4];
  const float* W1[4]; const float* b1[4]; const float* g1[4]; const float* bt1[4];
  const float* W2[4]; const float* b2[4]; const float* g2[4]; const float* bt2[4];
  const float* W3[4]; const float* b3[4];
  const float* self0; const float* self1;
  const int* edge_index;
  float* out; float* counts;
  h16* w1h[4]; h16* w2h[4]; h16* w3h[4];
  int E, N, in_feat;
};

// ---------------- fragment helpers ----------------
// Operands pre-shuffled to per-lane fragment order: frag[lane*16 + e]
// -> two contiguous b128 loads per lane.
// A inverse map (16-bit A 16x32, ISA 7.12.2): value (m,K): h=(K>>3)&1,
//   e=(K&7)+8*(K>>4), lane=m+16*h.
// B (32x16 KxN, same K order as A): lane=n+16*((K>>3)&1), e=(K&7)+8*(K>>4).
__device__ inline v16h load_frag(const h16* p) { return *(const v16h*)p; }

__device__ inline v8f wmma_f16(v16h a, v16h b, v8f c) {
  return __builtin_amdgcn_wmma_f32_16x16x32_f16(false, a, false, b, (short)0, c,
                                                false, false);
}

// async global -> LDS, 16B per lane (GVS mode: SGPR base + 32-bit VGPR offset)
__device__ inline void async_ld16(unsigned lds_off, unsigned voff, const void* sbase) {
  asm volatile("global_load_async_to_lds_b128 %0, %1, %2"
               :: "v"(lds_off), "v"(voff), "s"(sbase)
               : "memory");
}
__device__ inline void async_wait0() {
  asm volatile("s_wait_asynccnt 0x0" ::: "memory");
}

// LN over 32 feats/row then ReLU; writes f16 in A-fragment order.
__device__ inline void ln_relu_frag(const float* Yf, h16* Ya, float* mu, float* inv,
                                    const float* g, const float* bt, int tid) {
  if (tid < 16) {
    float s = 0.f;
    for (int c = 0; c < 32; ++c) s += Yf[tid * 32 + c];
    float m = s * (1.f / 32.f);
    float v = 0.f;
    for (int c = 0; c < 32; ++c) { float d = Yf[tid * 32 + c] - m; v += d * d; }
    mu[tid]  = m;
    inv[tid] = rsqrtf(v * (1.f / 32.f) + EPS);
  }
  __syncthreads();
  for (int idx = tid; idx < 512; idx += 128) {
    int m = idx >> 5, c = idx & 31;
    float y = (Yf[idx] - mu[m]) * inv[m] * g[c] + bt[c];
    int h = (c >> 3) & 1;
    int e = (c & 7) + 8 * (c >> 4);
    Ya[(m + 16 * h) * 16 + e] = (h16)fmaxf(y, 0.f);
  }
}

__global__ void k_zero(float* out, int n_out, float* counts, int n_cnt) {
  for (int i = blockIdx.x * blockDim.x + threadIdx.x; i < n_out + n_cnt;
       i += gridDim.x * blockDim.x) {
    if (i < n_out) out[i] = 0.f; else counts[i - n_out] = 0.f;
  }
}

// Weights -> f16, pre-shuffled into fragment order.
__global__ void k_convert(Args A) {
  int pi = blockIdx.y;
  int nf = ((pi >> 1) & (pi & 1)) ? 3 : 1;
  int out3 = 256 * nf;
  int total = 2048 + 1024 + 32 * out3;
  for (int idx = blockIdx.x * blockDim.x + threadIdx.x; idx < total;
       idx += gridDim.x * blockDim.x) {
    if (idx < 2048) {                 // W1 [33,32] -> 4 frags (kt,nt), K padded
      int f = idx >> 9, rem = idx & 511;
      int l = rem >> 4, e = rem & 15;
      int kt = f >> 1, nt = f & 1;
      int half = l >> 4;
      int n = nt * 16 + (l & 15);
      int k = kt * 32 + 8 * half + (e & 7) + 16 * (e >> 3);
      A.w1h[pi][idx] = (h16)((k < A.in_feat) ? A.W1[pi][k * 32 + n] : 0.f);
    } else if (idx < 3072) {          // W2 [32,32] -> 2 frags (nt)
      int i2 = idx - 2048;
      int f = i2 >> 9, rem = i2 & 511;
      int l = rem >> 4, e = rem & 15;
      int half = l >> 4;
      int n = f * 16 + (l & 15);
      int k = 8 * half + (e & 7) + 16 * (e >> 3);
      A.w2h[pi][i2] = (h16)A.W2[pi][k * 32 + n];
    } else {                          // W3 [32,out3] -> (out3/16) frags (t)
      int i3 = idx - 3072;
      int t = i3 >> 9, rem = i3 & 511;
      int l = rem >> 4, e = rem & 15;
      int half = l >> 4;
      int n = t * 16 + (l & 15);
      int k = 8 * half + (e & 7) + 16 * (e >> 3);
      A.w3h[pi][i3] = (h16)A.W3[pi][(size_t)k * out3 + n];
    }
  }
}

__global__ void k_count(const int* edge_index, float* counts, int E) {
  int i = blockIdx.x * blockDim.x + threadIdx.x;
  if (i < E) atomicAdd(&counts[edge_index[E + i]], 1.0f);
}

template <int PI>
__global__ __launch_bounds__(128) void k_main(Args A) {
  constexpr int DI   = PI >> 1, DD = PI & 1;
  constexpr int NF   = (DI & DD) ? 3 : 1;
  constexpr int OUT3 = 256 * NF;
  constexpr int P    = 2 * DD + 1, Q = 2 * DI + 1;
  constexpr int BSZ  = P * Q * NF;
  constexpr int W    = 16 * Q;            // src floats per edge
  constexpr int NT   = OUT3 >> 4;         // layer-3 N-tiles (16 or 48)

  const int e0 = blockIdx.x * 16;
  const int tid = threadIdx.x;
  const int lane = tid & 31, wave = tid >> 5;
  const int E = A.E;

  __shared__ __align__(16) float sFeatF32[16 * 32];
  __shared__ float sRf[16];
  __shared__ __align__(32) h16   sFeatA[1024];   // 2 K-tiles, fragment order
  __shared__ float sY[512];
  __shared__ __align__(32) h16   sYA[512];       // 1 K-tile, fragment order
  __shared__ float sMu[16], sInv[16];
  __shared__ float sR[16 * OUT3];
  __shared__ __align__(16) float sS[16 * W];
  __shared__ __align__(16) float sB[16 * BSZ];
  __shared__ int   sCol[16];

  __builtin_prefetch(A.w3h[PI], 0, 1);

  // ---- Stage 0: staging ----
  const bool full = (e0 + 16) <= E;
  if (full) {
    // edge_feat tile: 16*32 f32 = 2048B = 128 b128 chunks (one per thread)
    async_ld16((unsigned)(size_t)&sFeatF32[0] + tid * 16, tid * 16,
               A.edge_feat + (size_t)e0 * 32);
    // basis tile: contiguous 16*BSZ f32; chunks = BSZ*4 (4/12/12/108)
    if (tid < BSZ * 4)
      async_ld16((unsigned)(size_t)&sB[0] + tid * 16, tid * 16,
                 A.basis[PI] + (size_t)e0 * BSZ);
  } else {
    for (int idx = tid; idx < 16 * 32; idx += 128) {
      int er = idx >> 5, c = idx & 31;
      int e = e0 + er;
      sFeatF32[idx] = (e < E) ? A.edge_feat[(size_t)e * 32 + c] : 0.f;
    }
    for (int idx = tid; idx < 16 * BSZ; idx += 128) {
      int er = idx / BSZ, c = idx - er * BSZ;
      int e = e0 + er;
      sB[idx] = (e < E) ? A.basis[PI][(size_t)e * BSZ + c] : 0.f;
    }
  }
  if (tid < 16) {
    int e = e0 + tid;
    sRf[tid]  = (e < E) ? A.r[e] : 0.f;
    sCol[tid] = (e < E) ? A.edge_index[E + e] : -1;
  }
  {
    const float* hs = DI ? A.h1 : A.h0;    // gather source node features
    constexpr int CPE = W >> 2;            // float4 chunks per edge
    for (int idx = tid; idx < 16 * CPE; idx += 128) {
      int er = idx / CPE, c = idx - er * CPE;
      int e = e0 + er;
      float4 v = make_float4(0.f, 0.f, 0.f, 0.f);
      if (e < E) {
        int rw = A.edge_index[e];
        v = *(const float4*)(hs + (size_t)rw * W + c * 4);
      }
      *(float4*)(&sS[er * W + c * 4]) = v;
    }
  }
  if (full) async_wait0();
  __syncthreads();

  // feat (padded 33->64) -> A-fragment order, f16
  for (int idx = tid; idx < 16 * 64; idx += 128) {
    int m = idx >> 6, K = idx & 63;
    float v = 0.f;
    if (K < 32)       v = sFeatF32[m * 32 + K];
    else if (K == 32) v = sRf[m];
    int kt = K >> 5, Kl = K & 31;
    int h = (Kl >> 3) & 1;
    int e = (Kl & 7) + 8 * (Kl >> 4);
    sFeatA[kt * 512 + (m + 16 * h) * 16 + e] = (h16)v;
  }
  __syncthreads();

  // ---- Layer 1: [16,64]x[64,32] ----
  if (wave < 2) {
    v8f acc = {};
#pragma unroll
    for (int kt = 0; kt < 2; ++kt) {
      v16h a = load_frag(&sFeatA[kt * 512 + lane * 16]);
      v16h b = load_frag(A.w1h[PI] + (kt * 2 + wave) * 512 + lane * 16);
      acc = wmma_f16(a, b, acc);
    }
    int n = lane & 15, half = lane >> 4;
    const float* bias = A.b1[PI];
#pragma unroll
    for (int v = 0; v < 8; ++v)
      sY[(v + 8 * half) * 32 + wave * 16 + n] = acc[v] + bias[wave * 16 + n];
  }
  __syncthreads();
  ln_relu_frag(sY, sYA, sMu, sInv, A.g1[PI], A.bt1[PI], tid);
  __syncthreads();

  // ---- Layer 2: [16,32]x[32,32] ----
  if (wave < 2) {
    v16h a = load_frag(&sYA[lane * 16]);
    v16h b = load_frag(A.w2h[PI] + wave * 512 + lane * 16);
    v8f acc = {};
    acc = wmma_f16(a, b, acc);
    int n = lane & 15, half = lane >> 4;
    const float* bias = A.b2[PI];
#pragma unroll
    for (int v = 0; v < 8; ++v)
      sY[(v + 8 * half) * 32 + wave * 16 + n] = acc[v] + bias[wave * 16 + n];
  }
  __syncthreads();
  ln_relu_frag(sY, sYA, sMu, sInv, A.g2[PI], A.bt2[PI], tid);
  __syncthreads();

  // ---- Layer 3: [16,32]x[32,OUT3] -> R tile in LDS (fully unrolled) ----
  {
    v16h a = load_frag(&sYA[lane * 16]);
    const float* b3 = A.b3[PI];
    int n = lane & 15, half = lane >> 4;
#pragma unroll
    for (int tt = 0; tt < NT / 4; ++tt) {
      int t = tt * 4 + wave;
      v16h b = load_frag(A.w3h[PI] + t * 512 + lane * 16);
      v8f acc = {};
      acc = wmma_f16(a, b, acc);
#pragma unroll
      for (int v = 0; v < 8; ++v)
        sR[(v + 8 * half) * OUT3 + t * 16 + n] = acc[v] + b3[t * 16 + n];
    }
  }
  __syncthreads();

  // ---- Stage 4: fused basis contraction + message, atomic scatter ----
  {
    constexpr int TOT = 16 * 16 * P;
#pragma unroll 1
    for (int idx = tid; idx < TOT; idx += 128) {
      int e   = idx / (16 * P);
      int rem = idx - e * (16 * P);
      int o = rem / P, p = rem - o * P;
      int col = sCol[e];
      if (col < 0) continue;
      const float* Re = &sR[e * OUT3];
      const float* Be = &sB[e * BSZ];
      const float* Se = &sS[e * W];
      float acc = 0.f;
      for (int i = 0; i < 16; ++i) {
        const float* rr = &Re[(o * 16 + i) * NF];
#pragma unroll
        for (int q = 0; q < Q; ++q) {
          float s = Se[i * Q + q];
          const float* bb = &Be[(p * Q + q) * NF];
#pragma unroll
          for (int n = 0; n < NF; ++n) acc += rr[n] * bb[n] * s;
        }
      }
      float* dst = DD ? (A.out + (size_t)A.N * 16 + ((size_t)col * 16 + o) * 3 + p)
                      : (A.out + (size_t)col * 16 + o);
      atomicAdd(dst, acc);
    }
  }
}

__global__ void k_fin(Args A, int out_size) {
  int idx = blockIdx.x * blockDim.x + threadIdx.x;
  if (idx >= out_size) return;
  int sz0 = A.N * 16;
  float v;
  if (idx < sz0) {
    int n = idx >> 4, o = idx & 15;
    float cnt = A.counts[n];
    v = A.out[idx] / fmaxf(cnt, 1.f);
    if (cnt > 0.f) {
      const float* w = A.self0 + o * 16;
      const float* h = A.h0 + (size_t)n * 16;
      float s = 0.f;
      for (int i = 0; i < 16; ++i) s += w[i] * h[i];
      v += s;
    }
  } else {
    int j = idx - sz0;
    int n = j / 48; int rem = j - n * 48; int o = rem / 3, p = rem - o * 3;
    float cnt = A.counts[n];
    v = A.out[idx] / fmaxf(cnt, 1.f);
    if (cnt > 0.f) {
      const float* w = A.self1 + o * 16;
      const float* h = A.h1 + (size_t)n * 48 + p;
      float s = 0.f;
      for (int i = 0; i < 16; ++i) s += w[i] * h[i * 3];
      v += s;
    }
  }
  A.out[idx] = v;
}

extern "C" void kernel_launch(void* const* d_in, const int* in_sizes, int n_in,
                              void* d_out, int out_size, void* d_ws, size_t ws_size,
                              hipStream_t stream) {
  Args A;
  A.h0 = (const float*)d_in[0];
  A.h1 = (const float*)d_in[1];
  A.r  = (const float*)d_in[2];
  A.edge_feat = (const float*)d_in[3];
  for (int p = 0; p < 4; ++p) A.basis[p] = (const float*)d_in[4 + p];
  int b = 8;
  for (int p = 0; p < 4; ++p) {
    A.W1[p]  = (const float*)d_in[b + 0];
    A.b1[p]  = (const float*)d_in[b + 1];
    A.g1[p]  = (const float*)d_in[b + 2];
    A.bt1[p] = (const float*)d_in[b + 3];
    A.W2[p]  = (const float*)d_in[b + 4];
    A.b2[p]  = (const float*)d_in[b + 5];
    A.g2[p]  = (const float*)d_in[b + 6];
    A.bt2[p] = (const float*)d_in[b + 7];
    A.W3[p]  = (const float*)d_in[b + 8];
    A.b3[p]  = (const float*)d_in[b + 9];
    b += 10;
  }
  A.self0 = (const float*)d_in[48];
  A.self1 = (const float*)d_in[49];
  A.edge_index = (const int*)d_in[50];
  A.out = (float*)d_out;
  A.E = in_sizes[2];
  A.N = in_sizes[0] / 16;
  A.in_feat = in_sizes[3] / A.E + 1;   // EDGE_DIM + 1 = 33

  char* ws = (char*)d_ws;
  A.counts = (float*)ws;
  size_t off = ((size_t)A.N * 4 + 255) & ~(size_t)255;
  for (int p = 0; p < 4; ++p) { A.w1h[p] = (h16*)(ws + off); off += 2048 * 2; }
  for (int p = 0; p < 4; ++p) { A.w2h[p] = (h16*)(ws + off); off += 1024 * 2; }
  for (int p = 0; p < 4; ++p) {
    int nf = (p == 3) ? 3 : 1;
    A.w3h[p] = (h16*)(ws + off);
    off += (size_t)32 * 256 * nf * 2;
  }
  (void)ws_size; (void)n_in;

  int E = A.E, N = A.N;
  int totz = out_size + N;
  k_zero<<<(totz + 255) / 256, 256, 0, stream>>>(A.out, out_size, A.counts, N);
  k_convert<<<dim3(32, 4), 256, 0, stream>>>(A);
  k_count<<<(E + 255) / 256, 256, 0, stream>>>(A.edge_index, A.counts, E);
  int tiles = (E + 15) / 16;
  k_main<0><<<tiles, 128, 0, stream>>>(A);
  k_main<1><<<tiles, 128, 0, stream>>>(A);
  k_main<2><<<tiles, 128, 0, stream>>>(A);
  k_main<3><<<tiles, 128, 0, stream>>>(A);
  k_fin<<<(out_size + 255) / 256, 256, 0, stream>>>(A, out_size);
}
